// LongConvModel_4080218931784
// MI455X (gfx1250) — matrix operand order repaired and checked by reading, not compiled, not run on hardware
//
#include <hip/hip_runtime.h>

// ---------------------------------------------------------------------------
// LongConvModel forward for MI455X (gfx1250, wave32, WMMA + async-to-LDS)
//   B=32, L=1024, Di=128, H=512, C=2, 2L=2048, layers=6, out=(32,10) fp32
// ---------------------------------------------------------------------------

#define B_   32
#define L_   1024
#define DI_  128
#define H_   512
#define C_   2
#define N2L  2048
#define LOG2L 11
#define NLAYERS 6

typedef __attribute__((ext_vector_type(16))) __bf16 v16bf;
typedef __attribute__((ext_vector_type(8)))  __bf16 v8bf;
typedef __attribute__((ext_vector_type(8)))  float  v8f;
typedef int v4i __attribute__((vector_size(16)));   // matches builtin param pointee

#if defined(__gfx1250__) && __has_builtin(__builtin_amdgcn_global_load_async_to_lds_b128)
#define HAVE_ASYNC_LDS 1
#else
#define HAVE_ASYNC_LDS 0
#endif

#if HAVE_ASYNC_LDS
__device__ __forceinline__ void async_copy_b128_to_lds(const void* gsrc, void* ldst) {
  __builtin_amdgcn_global_load_async_to_lds_b128(
      (__attribute__((address_space(1))) v4i*)(gsrc),
      (__attribute__((address_space(3))) v4i*)(ldst), 0, 0);
}
__device__ __forceinline__ void wait_async_lds() {
#if __has_builtin(__builtin_amdgcn_s_wait_asynccnt)
  __builtin_amdgcn_s_wait_asynccnt(0);
#else
  asm volatile("s_wait_asynccnt 0" ::: "memory");
#endif
}
#endif

// ------------------------------ small helpers ------------------------------

__device__ __forceinline__ float2 cmul(float2 a, float2 b) {
  return make_float2(a.x * b.x - a.y * b.y, a.x * b.y + a.y * b.x);
}

// ------------------------------ FFT (radix-2) ------------------------------
// Forward: DIF, natural input -> bit-reversed output, twiddle e^{-i...}
// Inverse: DIT, bit-reversed input -> natural output, twiddle e^{+i...}
// Pointwise product in bit-reversed order is order-consistent, so no
// bit-reversal permutation is ever materialized.

__device__ void fft2048_dif(float2* s, int tid, int nth) {
  for (int st = LOG2L; st >= 1; --st) {
    int half = 1 << (st - 1);
    for (int p = tid; p < (N2L / 2); p += nth) {
      int j  = p & (half - 1);
      int i0 = ((p >> (st - 1)) << st) + j;
      int i1 = i0 + half;
      float2 a = s[i0], b = s[i1];
      float2 sum = make_float2(a.x + b.x, a.y + b.y);
      float2 dif = make_float2(a.x - b.x, a.y - b.y);
      float ang = -3.14159265358979f * (float)j / (float)half;
      float sn, cs; __sincosf(ang, &sn, &cs);
      s[i0] = sum;
      s[i1] = cmul(dif, make_float2(cs, sn));
    }
    __syncthreads();
  }
}

__device__ void fft2048_dit_inv(float2* s, int tid, int nth) {
  for (int st = 1; st <= LOG2L; ++st) {
    int half = 1 << (st - 1);
    for (int p = tid; p < (N2L / 2); p += nth) {
      int j  = p & (half - 1);
      int i0 = ((p >> (st - 1)) << st) + j;
      int i1 = i0 + half;
      float ang = 3.14159265358979f * (float)j / (float)half;
      float sn, cs; __sincosf(ang, &sn, &cs);
      float2 t = cmul(s[i1], make_float2(cs, sn));
      float2 a = s[i0];
      s[i0] = make_float2(a.x + t.x, a.y + t.y);
      s[i1] = make_float2(a.x - t.x, a.y - t.y);
    }
    __syncthreads();
  }
}

// -------------------- kernel spectrum: squash + forward FFT -----------------
// one block per (c,h); kern layer slice is (C,H,2L); kf out (C*H, 2048) cplx
__global__ __launch_bounds__(256) void kernel_fft_k(const float* __restrict__ kern,
                                                    float2* __restrict__ kf) {
  __shared__ float2 S[N2L];
  int ch = blockIdx.x;  // c*H + h
  const float* kp = kern + (size_t)ch * N2L;
  for (int l = threadIdx.x; l < N2L; l += blockDim.x) {
    float v = kp[l];
    float av = fabsf(v) - 0.1f;               // KERNEL_LAM
    float sq = (av > 0.f) ? ((v > 0.f) ? av : -av) : 0.f;
    S[l] = make_float2(sq, 0.f);
  }
  __syncthreads();
  fft2048_dif(S, threadIdx.x, blockDim.x);
  float2* outp = kf + (size_t)ch * N2L;
  for (int l = threadIdx.x; l < N2L; l += blockDim.x) outp[l] = S[l];
}

// -------- FFT convolution: y[b, c*H+h, l] = (k *circ u)[l] + D[c,h]*u[l] ----
// one block per (h, b). 32KB LDS.
__global__ __launch_bounds__(256) void conv_fft_k(const float* __restrict__ h,
                                                  const float2* __restrict__ kf,
                                                  const float* __restrict__ Dp,
                                                  float* __restrict__ yconv) {
  __shared__ float2 U[N2L];
  __shared__ float2 V[N2L];
  int hh = blockIdx.x;   // 0..511
  int b  = blockIdx.y;   // 0..31
  const float* up = h + ((size_t)b * H_ + hh) * L_;
  for (int l = threadIdx.x; l < L_; l += blockDim.x) {
    U[l]      = make_float2(up[l], 0.f);
    U[l + L_] = make_float2(0.f, 0.f);
  }
  __syncthreads();
  fft2048_dif(U, threadIdx.x, blockDim.x);
  for (int c = 0; c < C_; ++c) {
    const float2* kp = kf + ((size_t)c * H_ + hh) * N2L;
    for (int i = threadIdx.x; i < N2L; i += blockDim.x) V[i] = cmul(U[i], kp[i]);
    __syncthreads();
    fft2048_dit_inv(V, threadIdx.x, blockDim.x);
    float dd = Dp[c * H_ + hh];
    float* yp = yconv + ((size_t)b * (C_ * H_) + c * H_ + hh) * L_;
    for (int l = threadIdx.x; l < L_; l += blockDim.x)
      yp[l] = V[l].x * (1.0f / (float)N2L) + dd * up[l];
    __syncthreads();  // V reused next channel
  }
}

// -------------- transpose (B,2H,L)->(B*L,2H) with fused tanh-GELU -----------
__global__ __launch_bounds__(256) void transpose_gelu_k(const float* __restrict__ y,
                                                        __bf16* __restrict__ X) {
  __shared__ float T[32][33];
  int l0 = blockIdx.x * 32, h0 = blockIdx.y * 32, b = blockIdx.z;
  int tx = threadIdx.x, ty = threadIdx.y;          // 32 x 8
  for (int j = 0; j < 4; ++j) {
    int hh = h0 + ty + j * 8;
    T[ty + j * 8][tx] = y[((size_t)b * (C_ * H_) + hh) * L_ + l0 + tx];
  }
  __syncthreads();
  for (int j = 0; j < 4; ++j) {
    int l = l0 + ty + j * 8;
    float v = T[tx][ty + j * 8];
    float g = 0.5f * v * (1.f + tanhf(0.7978845608028654f * (v + 0.044715f * v * v * v)));
    X[((size_t)b * L_ + l) * (C_ * H_) + h0 + tx] = (__bf16)g;
  }
}

// ------------------------------ f32 -> bf16 --------------------------------
__global__ __launch_bounds__(256) void f32_to_bf16_k(const float* __restrict__ in,
                                                     __bf16* __restrict__ out, int n) {
  int i = blockIdx.x * blockDim.x + threadIdx.x;
  if (i < n) out[i] = (__bf16)in[i];
}

// --------------------------- WMMA bf16 GEMM --------------------------------
// out(M,N) = A(M,K)bf16 @ W(K,N)bf16 + bias(N), fp32 accumulate.
// MODE 0: out[m*N+n]   MODE 1: out[(m/L)*N*L + n*L + (m%L)]  (write (B,N,L))
// block tile 128x64, 8 waves, each wave 32x32 (2x2 WMMA frags), BK=32.
// A tile is staged straight into LDS with GLOBAL_LOAD_ASYNC_TO_LDS_B128 when
// the toolchain exposes it (ASYNCcnt + s_wait_asynccnt), else via VGPR bounce.
#define BM 128
#define BN 64
#define BK 32
#define LDS_STRIDE 40   // BK + 8 pad, keeps 16B alignment, conflict-free b128

template <int MODE>
__global__ __launch_bounds__(256) void gemm_bf16_wmma_k(
    const __bf16* __restrict__ A, const __bf16* __restrict__ W,
    const float* __restrict__ bias, float* __restrict__ out,
    int M, int N, int K) {
  __shared__ __align__(16) __bf16 As[BM * LDS_STRIDE];  // 10 KB
  __shared__ __align__(16) __bf16 Ws[BN * LDS_STRIDE];  // 5 KB  (stored W^T)
  int tid  = threadIdx.x;
  int lane = tid & 31;
  int wave = tid >> 5;
  int wm = wave & 3;           // 4 M-subtiles of 32
  int wn = wave >> 2;          // 2 N-subtiles of 32
  int m0 = blockIdx.y * BM;
  int n0 = blockIdx.x * BN;

  v8f acc[2][2] = {};

  int frow = lane & 15;                       // fragment row/col within 16
  int kba  = (lane < 16) ? 0 : 8;             // A frag K base (ISA 16-bit A layout)
  int kbb  = (lane < 16) ? 0 : 8;             // B frag K base

  for (int k0 = 0; k0 < K; k0 += BK) {
    // --- stage A tile 128x32 (16B chunks, global -> LDS) ---
    {
      int chunk = tid & 3;
      int r0 = tid >> 2;       // 0..63
      for (int r = r0; r < BM; r += 64) {
        const __bf16* gsrc = A + (size_t)(m0 + r) * K + k0 + chunk * 8;
        __bf16* ldst = &As[r * LDS_STRIDE + chunk * 8];
#if HAVE_ASYNC_LDS
        async_copy_b128_to_lds(gsrc, ldst);
#else
        *(v8bf*)ldst = *(const v8bf*)gsrc;
#endif
        if (k0 + BK < K)
          __builtin_prefetch(gsrc + BK, 0, 3);
      }
    }
    // --- stage W tile 32x64 transposed into Ws[n][k] ---
    {
      int n  = tid & 63;
      int kk = tid >> 6;       // 0..3
      for (int k2 = kk; k2 < BK; k2 += 4)
        Ws[n * LDS_STRIDE + k2] = W[(size_t)(k0 + k2) * N + n0 + n];
    }
#if HAVE_ASYNC_LDS
    wait_async_lds();
#endif
    __syncthreads();

    v16bf afrag[2], bfrag[2];
    for (int t = 0; t < 2; ++t) {
      const __bf16* ap = &As[(wm * 32 + t * 16 + frow) * LDS_STRIDE + kba];
      v8bf alo = *(const v8bf*)(ap);
      v8bf ahi = *(const v8bf*)(ap + 16);
      afrag[t] = __builtin_shufflevector(alo, ahi, 0, 1, 2, 3, 4, 5, 6, 7,
                                         8, 9, 10, 11, 12, 13, 14, 15);
      const __bf16* bp = &Ws[(wn * 32 + t * 16 + frow) * LDS_STRIDE + kbb];
      v8bf blo = *(const v8bf*)(bp);
      v8bf bhi = *(const v8bf*)(bp + 16);
      bfrag[t] = __builtin_shufflevector(blo, bhi, 0, 1, 2, 3, 4, 5, 6, 7,
                                         8, 9, 10, 11, 12, 13, 14, 15);
    }
    for (int i = 0; i < 2; ++i)
      for (int j = 0; j < 2; ++j)
        acc[i][j] = __builtin_amdgcn_wmma_f32_16x16x32_bf16(
            false, afrag[i], false, bfrag[j], (short)0, acc[i][j], false, false);
    __syncthreads();
  }

  // --- epilogue: C/D layout (8 VGPRs; M = r + 8*lane[4], N = lane&15) ---
  int col   = lane & 15;
  int rbase = (lane >> 4) << 3;
  for (int i = 0; i < 2; ++i) {
    for (int j = 0; j < 2; ++j) {
      int n = n0 + wn * 32 + j * 16 + col;
      float bv = bias[n];
      for (int r = 0; r < 8; ++r) {
        int m = m0 + wm * 32 + i * 16 + rbase + r;
        float v = acc[i][j][r] + bv;
        if (MODE == 0) {
          out[(size_t)m * N + n] = v;
        } else {
          int bb = m >> 10, l = m & (L_ - 1);
          out[((size_t)bb * N + n) * L_ + l] = v;
        }
      }
    }
  }
}

// -------------------- GLU + residual: h[b,n,l] += a*sigmoid(g) --------------
__global__ __launch_bounds__(256) void glu_residual_k(const float* __restrict__ yd,
                                                      float* __restrict__ h) {
  size_t idx = (size_t)blockIdx.x * blockDim.x + threadIdx.x;  // B*H*L
  if (idx >= (size_t)B_ * H_ * L_) return;
  int l = idx & (L_ - 1);
  int n = (int)((idx >> 10) & (H_ - 1));
  int b = (int)(idx >> 19);
  size_t m = (size_t)b * L_ + l;
  float a = yd[m * (2 * H_) + n];
  float g = yd[m * (2 * H_) + H_ + n];
  h[idx] += a * (1.f / (1.f + __expf(-g)));
}

// ----------------- LayerNorm over L with per-position scale/bias ------------
__global__ __launch_bounds__(256) void layernorm_k(float* __restrict__ h,
                                                   const float* __restrict__ scale,
                                                   const float* __restrict__ bias) {
  __shared__ float r1[256];
  __shared__ float r2[256];
  int row = blockIdx.x;                        // b*H + hh
  float* x = h + (size_t)row * L_;
  float vals[4];
  float s = 0.f, s2 = 0.f;
  for (int j = 0; j < 4; ++j) {
    float v = x[threadIdx.x + j * 256];
    vals[j] = v; s += v; s2 += v * v;
  }
  r1[threadIdx.x] = s; r2[threadIdx.x] = s2;
  __syncthreads();
  for (int st = 128; st > 0; st >>= 1) {
    if (threadIdx.x < st) { r1[threadIdx.x] += r1[threadIdx.x + st];
                            r2[threadIdx.x] += r2[threadIdx.x + st]; }
    __syncthreads();
  }
  float mu  = r1[0] * (1.f / (float)L_);
  float var = r2[0] * (1.f / (float)L_) - mu * mu;
  float rs  = rsqrtf(var + 1e-6f);
  for (int j = 0; j < 4; ++j) {
    int l = threadIdx.x + j * 256;
    x[l] = (vals[j] - mu) * rs * scale[l] + bias[l];
  }
}

// ------------------------------ mean over L --------------------------------
__global__ __launch_bounds__(256) void mean_k(const float* __restrict__ h,
                                              float* __restrict__ hbar) {
  __shared__ float red[256];
  int row = blockIdx.x;
  const float* x = h + (size_t)row * L_;
  float s = 0.f;
  for (int j = 0; j < 4; ++j) s += x[threadIdx.x + j * 256];
  red[threadIdx.x] = s;
  __syncthreads();
  for (int st = 128; st > 0; st >>= 1) {
    if (threadIdx.x < st) red[threadIdx.x] += red[threadIdx.x + st];
    __syncthreads();
  }
  if (threadIdx.x == 0) hbar[row] = red[0] * (1.f / (float)L_);
}

// ------------------------------- decoder -----------------------------------
__global__ void decode_k(const float* __restrict__ hbar, const float* __restrict__ w,
                         const float* __restrict__ bvec, float* __restrict__ out) {
  int t = threadIdx.x;       // 320 threads: (b, o)
  if (t >= B_ * 10) return;
  int b = t / 10, o = t % 10;
  float s = bvec[o];
  for (int k = 0; k < H_; ++k) s += hbar[b * H_ + k] * w[k * 10 + o];
  out[t] = s;
}

// ---------------------------------------------------------------------------

extern "C" void kernel_launch(void* const* d_in, const int* in_sizes, int n_in,
                              void* d_out, int out_size, void* d_ws, size_t ws_size,
                              hipStream_t stream) {
  (void)in_sizes; (void)n_in; (void)out_size; (void)ws_size;
  const float* x       = (const float*)d_in[0];   // (32,1024,128)
  const float* enc_w   = (const float*)d_in[1];   // (128,512)
  const float* enc_b   = (const float*)d_in[2];   // (512,)
  const float* Dmat    = (const float*)d_in[3];   // (6,2,512)
  const float* kern    = (const float*)d_in[4];   // (6,2,512,2048)
  const float* dense_w = (const float*)d_in[5];   // (6,1024,1024)
  const float* dense_b = (const float*)d_in[6];   // (6,1024)
  const float* ln_s    = (const float*)d_in[7];   // (6,1024)
  const float* ln_b    = (const float*)d_in[8];   // (6,1024)
  const float* dec_w   = (const float*)d_in[9];   // (512,10)
  const float* dec_b   = (const float*)d_in[10];  // (10,)
  float* out = (float*)d_out;

  char* ws = (char*)d_ws;
  size_t off = 0;
  auto carve = [&](size_t bytes) -> void* {
    void* p = ws + off;
    off = (off + bytes + 255) & ~(size_t)255;
    return p;
  };
  float*  h     = (float*) carve((size_t)B_ * H_ * L_ * 4);            //  64 MB
  float2* kf    = (float2*)carve((size_t)C_ * H_ * N2L * 8);           //  16 MB
  float*  yconv = (float*) carve((size_t)B_ * C_ * H_ * L_ * 4);       // 128 MB
  __bf16* Xbf   = (__bf16*)carve((size_t)B_ * L_ * C_ * H_ * 2);       //  64 MB
  __bf16* wbf   = (__bf16*)carve((size_t)(C_ * H_) * (2 * H_) * 2);    //   2 MB
  float*  hbar  = (float*) carve((size_t)B_ * H_ * 4);
  float*  yd = yconv;   // GEMM output aliases yconv (conv buffer dead by then)

  const int M = B_ * L_;  // 32768

  // ---- encoder: h(B,H,L) = transpose(x @ enc_w + enc_b) ----
  {
    int nx = B_ * L_ * DI_;
    f32_to_bf16_k<<<(nx + 255) / 256, 256, 0, stream>>>(x, Xbf, nx);
    int nw = DI_ * H_;
    f32_to_bf16_k<<<(nw + 255) / 256, 256, 0, stream>>>(enc_w, wbf, nw);
    dim3 g(H_ / BN, M / BM);
    gemm_bf16_wmma_k<1><<<g, 256, 0, stream>>>(Xbf, wbf, enc_b, h, M, H_, DI_);
  }

  for (int i = 0; i < NLAYERS; ++i) {
    kernel_fft_k<<<C_ * H_, 256, 0, stream>>>(kern + (size_t)i * C_ * H_ * N2L, kf);
    conv_fft_k<<<dim3(H_, B_), 256, 0, stream>>>(h, kf, Dmat + (size_t)i * C_ * H_, yconv);
    transpose_gelu_k<<<dim3(L_ / 32, (C_ * H_) / 32, B_), dim3(32, 8), 0, stream>>>(yconv, Xbf);
    int nw = (C_ * H_) * (2 * H_);
    f32_to_bf16_k<<<(nw + 255) / 256, 256, 0, stream>>>(dense_w + (size_t)i * nw, wbf, nw);
    dim3 g((2 * H_) / BN, M / BM);
    gemm_bf16_wmma_k<0><<<g, 256, 0, stream>>>(Xbf, wbf, dense_b + (size_t)i * 2 * H_,
                                               yd, M, 2 * H_, C_ * H_);
    glu_residual_k<<<(B_ * H_ * L_) / 256, 256, 0, stream>>>(yd, h);
    layernorm_k<<<B_ * H_, 256, 0, stream>>>(h, ln_s + (size_t)i * L_, ln_b + (size_t)i * L_);
  }

  mean_k<<<B_ * H_, 256, 0, stream>>>(h, hbar);
  decode_k<<<1, 320, 0, stream>>>(hbar, dec_w, dec_b, out);
}
